// BinaryDense_28054726378309
// MI455X (gfx1250) — matrix operand order repaired and checked by reading, not compiled, not run on hardware
//
#include <hip/hip_runtime.h>

typedef __attribute__((ext_vector_type(16))) __bf16 v16bf;
typedef __attribute__((ext_vector_type(8)))  float  v8f;
typedef __attribute__((ext_vector_type(4)))  float  v4f;

__device__ __forceinline__ unsigned rotl32(unsigned x, int r) {
    return (x << r) | (x >> (32 - r));
}

// ---------------------------------------------------------------------------
// Kernel 1: binarize the 128x128 weight matrix.
// Reproduces JAX: p = clip((w+1)*0.5, 0, 1); u = threefry2x32 uniform;
// bin = (u < p) ? +1 : -1   (training)  /  sign(w) (inference).
// Writes +-1 as bf16 directly in the WMMA B-fragment layout:
//   frag[(nt*4 + ks)*32 + lane][j], lane = (kr>>4)*16 + (f&15), j = kr&15
// so the GEMM kernel can load each lane's 32 bytes contiguously.
// ---------------------------------------------------------------------------
__global__ void binarize_weights(const float* __restrict__ w,
                                 const int*  __restrict__ seed_p,
                                 const int*  __restrict__ train_p,
                                 __bf16*     __restrict__ bfrag) {
    int idx = blockIdx.x * blockDim.x + threadIdx.x;   // 0..16383
    if (idx >= 128 * 128) return;
    int d = idx >> 7;    // K index (row of weight)
    int f = idx & 127;   // N index (col of weight)
    float wv = w[idx];
    float bin;
    if (*train_p) {
        long long s = (long long)(*seed_p);
        unsigned k0 = (unsigned)(((unsigned long long)s) >> 32);
        unsigned k1 = (unsigned)s;
        const unsigned half = 8192;
        unsigned i0 = (idx < (int)half) ? (unsigned)idx : (unsigned)(idx - half);
        unsigned x0 = i0, x1 = i0 + half;
        unsigned kk0 = k0, kk1 = k1, kk2 = k0 ^ k1 ^ 0x1BD11BDAu;
        unsigned kk[3] = {kk0, kk1, kk2};
        x0 += kk0; x1 += kk1;
        const int rot[2][4] = {{13, 15, 26, 6}, {17, 29, 16, 24}};
#pragma unroll
        for (int i = 0; i < 5; ++i) {
#pragma unroll
            for (int j = 0; j < 4; ++j) {
                x0 += x1;
                x1 = rotl32(x1, rot[i & 1][j]);
                x1 ^= x0;
            }
            x0 += kk[(i + 1) % 3];
            x1 += kk[(i + 2) % 3] + (unsigned)(i + 1);
        }
        unsigned bits = (idx < (int)half) ? x0 : x1;
        float u = __uint_as_float((bits >> 9) | 0x3f800000u) - 1.0f;
        float p = fminf(fmaxf((wv + 1.0f) * 0.5f, 0.0f), 1.0f);
        bin = (u < p) ? 1.0f : -1.0f;
    } else {
        bin = (wv > 0.0f) ? 1.0f : -1.0f;
    }
    int nt = f >> 4, ln = f & 15;
    int ks = d >> 5, kr = d & 31;
    int lane = ((kr >> 4) << 4) | ln;
    int j = kr & 15;
    bfrag[((nt * 4 + ks) * 32 + lane) * 16 + j] = (__bf16)bin;
}

// ---------------------------------------------------------------------------
// Kernel 2: C[N,128] = A[N,128] @ Wbin[128,128] via bf16 WMMA with hi/lo split
// for fp32 accuracy (Wbin is exactly representable in bf16).
// Each wave: 32 rows (2 M-tiles) x 128 cols. A fragments (hi+lo for all 4
// k-steps) live in registers; B fragments stream from L2-resident workspace.
// ---------------------------------------------------------------------------
__global__ __launch_bounds__(256) void gemm_bin_wmma(
    const float*  __restrict__ A,
    const __bf16* __restrict__ Bf,
    float*        __restrict__ C) {
    const int lane  = threadIdx.x & 31;
    const int wave  = threadIdx.x >> 5;
    const long long rowBase = (long long)blockIdx.x * 256 + (long long)wave * 32;
    const int mrow  = lane & 15;
    const int khalf = (lane >> 4) ? 16 : 0;  // A layout: lanes 16-31 hold K+16

    // Load A once, split into hi/lo bf16 fragments (kept in VGPRs).
    v16bf a_hi[2][4], a_lo[2][4];
#pragma unroll
    for (int mt = 0; mt < 2; ++mt) {
        const float* ap = A + (rowBase + mt * 16 + mrow) * 128;
#pragma unroll
        for (int ks = 0; ks < 4; ++ks) {
            const float* p = ap + ks * 32 + khalf;
            v4f q[4];
            q[0] = *(const v4f*)(p + 0);     // K = base+0..3
            q[1] = *(const v4f*)(p + 4);     // K = base+4..7
            q[2] = *(const v4f*)(p + 8);     // K = base+8..11
            q[3] = *(const v4f*)(p + 12);    // K = base+12..15
            v16bf hi, lo;
#pragma unroll
            for (int e = 0; e < 16; ++e) {
                float x = q[e >> 2][e & 3];
                __bf16 h = (__bf16)x;
                hi[e] = h;
                lo[e] = (__bf16)(x - (float)h);
            }
            a_hi[mt][ks] = hi;
            a_lo[mt][ks] = lo;
        }
    }

    for (int nt = 0; nt < 8; ++nt) {
        v8f acc0 = {};   // rows rowBase..+15
        v8f acc1 = {};   // rows rowBase+16..+31
#pragma unroll
        for (int ks = 0; ks < 4; ++ks) {
            v16bf b = *(const v16bf*)(Bf + ((nt * 4 + ks) * 32 + lane) * 16);
            acc0 = __builtin_amdgcn_wmma_f32_16x16x32_bf16(
                false, a_hi[0][ks], false, b, (short)0, acc0, false, false);
            acc0 = __builtin_amdgcn_wmma_f32_16x16x32_bf16(
                false, a_lo[0][ks], false, b, (short)0, acc0, false, false);
            acc1 = __builtin_amdgcn_wmma_f32_16x16x32_bf16(
                false, a_hi[1][ks], false, b, (short)0, acc1, false, false);
            acc1 = __builtin_amdgcn_wmma_f32_16x16x32_bf16(
                false, a_lo[1][ks], false, b, (short)0, acc1, false, false);
        }
        // C/D layout: VGPR v, lanes 0-15 -> M=v, lanes 16-31 -> M=8+v; N=lane&15
        const int col = nt * 16 + (lane & 15);
        const long long r0 = rowBase + (long long)((lane >> 4) * 8);
        float* c0 = C + r0 * 128 + col;
#pragma unroll
        for (int v = 0; v < 8; ++v) {
            c0[(long long)v * 128]        = acc0[v];
            c0[(long long)(v + 16) * 128] = acc1[v];
        }
    }
}

extern "C" void kernel_launch(void* const* d_in, const int* in_sizes, int n_in,
                              void* d_out, int out_size, void* d_ws, size_t ws_size,
                              hipStream_t stream) {
    const float* inputs = (const float*)d_in[0];
    const float* weight = (const float*)d_in[1];
    const int*   seed   = (const int*)d_in[2];
    const int*   train  = (const int*)d_in[3];
    float*       out    = (float*)d_out;
    __bf16*      bfrag  = (__bf16*)d_ws;   // 128*128 bf16 = 32 KB

    binarize_weights<<<64, 256, 0, stream>>>(weight, seed, train, bfrag);

    int n = in_sizes[0] / 128;             // number of rows (1048576)
    int grid = n / 256;                    // 256 rows per block (8 waves x 32)
    gemm_bin_wmma<<<grid, 256, 0, stream>>>(inputs, bfrag, out);
}
